// GraphConv_47064251629851
// MI455X (gfx1250) — matrix operand order repaired and checked by reading, not compiled, not run on hardware
//
#include <hip/hip_runtime.h>
#include <hip/hip_bf16.h>

typedef __attribute__((ext_vector_type(16))) __bf16 v16bf;
typedef __attribute__((ext_vector_type(8)))  float  v8f;

#define D_FEAT 64
#define D_OUT  64

// ---------------------------------------------------------------------------
// Kernel 1: zero a float buffer with b128 stores (agg + deg accumulators)
// ---------------------------------------------------------------------------
__global__ void gc_zero4_kernel(float4* __restrict__ p, long long n4) {
    long long i = (long long)blockIdx.x * blockDim.x + threadIdx.x;
    long long stride = (long long)gridDim.x * blockDim.x;
    const float4 z = {0.f, 0.f, 0.f, 0.f};
    for (; i < n4; i += stride) p[i] = z;
}

__global__ void gc_zero_kernel(float* __restrict__ p, long long n) {
    long long i = (long long)blockIdx.x * blockDim.x + threadIdx.x;
    long long stride = (long long)gridDim.x * blockDim.x;
    for (; i < n; i += stride) p[i] = 0.0f;
}

// ---------------------------------------------------------------------------
// Kernel 2: edge scatter. One wave (32 lanes) per edge; each lane moves a
// float2 slice of the 64-feature row (coalesced 256B gather, L2-resident)
// and does two native f32 atomics into agg[dst]. Lane 0 bumps deg[dst].
// ---------------------------------------------------------------------------
__global__ void __launch_bounds__(256)
gc_scatter_kernel(const float* __restrict__ vertices,
                  const long long* __restrict__ edges,
                  float* __restrict__ agg,
                  float* __restrict__ deg,
                  int n_edges) {
    long long t = (long long)blockIdx.x * blockDim.x + threadIdx.x;
    int e    = (int)(t >> 5);
    int lane = (int)(t & 31);
    if (e >= n_edges) return;

    long long dst = edges[e];                       // edges[0][e]
    long long src = edges[(long long)n_edges + e];  // edges[1][e]

    const float2 v = ((const float2*)(vertices + src * D_FEAT))[lane];
    float* arow = agg + dst * D_FEAT + lane * 2;
    unsafeAtomicAdd(arow,     v.x);
    unsafeAtomicAdd(arow + 1, v.y);
    if (lane == 0) unsafeAtomicAdd(deg + dst, 1.0f);
}

// ---------------------------------------------------------------------------
// Kernel 3: fused finalize + GEMM.
//   A[m][k] = (agg[m][k] + vertices[m][k]) * rsqrt(deg[m] + 1)   (bf16)
//   out     = A @ W^T + bias     via v_wmma_f32_16x16x32_bf16
// Block = 128 threads = 4 waves; block owns 16 node rows, wave w owns the
// 16-wide output column tile n0 = 16*w. Two WMMAs chain over K = 64.
// Bias is preloaded into the accumulator (C fragment).
// ---------------------------------------------------------------------------
__global__ void __launch_bounds__(128)
gc_gemm_kernel(const float* __restrict__ vertices,
               const float* __restrict__ weight,   // [D_OUT][D_FEAT] row-major
               const float* __restrict__ bias,     // [D_OUT]
               const float* __restrict__ agg,      // [n][64] (may alias out)
               const float* __restrict__ deg,      // [n]
               float* __restrict__ out,            // [n][64]
               int n_nodes) {
    const int m0   = blockIdx.x * 16;
    const int wave = threadIdx.x >> 5;      // 0..3 -> N tile
    const int lane = threadIdx.x & 31;
    const int col  = lane & 15;             // N (for B/C/D) / M (for A)
    const int hi   = lane >> 4;             // lane group selects K half

    const int n0 = wave * 16;

    // ---- A fragments: lane holds row m = m0+col; ISA 16-bit A 16x32 layout:
    //   lanes 0-15 : elems 0-7 -> K=0..7,   elems 8-15 -> K=16..23
    //   lanes16-31 : elems 0-7 -> K=8..15,  elems 8-15 -> K=24..31
    int m = m0 + col;
    if (m > n_nodes - 1) m = n_nodes - 1;   // clamp: keep EXEC all-ones for WMMA
    const float scale = rsqrtf(deg[m] + 1.0f);   // +1 self-loop; always >= 1
    const float* arow = agg      + (long long)m * D_FEAT;
    const float* vrow = vertices + (long long)m * D_FEAT;

    v16bf a0, a1, b0, b1;
    #pragma unroll
    for (int e = 0; e < 8; ++e) {
        int kA = hi * 8 + e;
        int kB = hi * 8 + 16 + e;
        a0[e]     = (__bf16)((arow[kA]      + vrow[kA])      * scale);
        a0[e + 8] = (__bf16)((arow[kB]      + vrow[kB])      * scale);
        a1[e]     = (__bf16)((arow[32 + kA] + vrow[32 + kA]) * scale);
        a1[e + 8] = (__bf16)((arow[32 + kB] + vrow[32 + kB]) * scale);
    }

    // ---- B fragments: B[k][n] = W[n][k]; 16-bit B 32x16 layout:
    //   lanes 0-15 : elems 0-7 -> K=0..7,  elems 8-15 -> K=16..23 of column n=col
    //   lanes16-31 : elems 0-7 -> K=8..15, elems 8-15 -> K=24..31
    const float* wrow = weight + (long long)(n0 + col) * D_FEAT + hi * 8;
    #pragma unroll
    for (int e = 0; e < 16; ++e) {
        b0[e] = (__bf16)wrow[e + (e & 8)];        // K chunk 0 (features 0..31)
        b1[e] = (__bf16)wrow[32 + e + (e & 8)];   // K chunk 1 (features 32..63)
    }

    // All loads done before any store: makes agg==out aliasing safe.
    __syncthreads();

    // ---- accumulator preloaded with bias (C fragment: N = col for all rows)
    const float bv = bias[n0 + col];
    v8f acc;
    #pragma unroll
    for (int r = 0; r < 8; ++r) acc[r] = bv;

    acc = __builtin_amdgcn_wmma_f32_16x16x32_bf16(false, a0, false, b0,
                                                  (short)0, acc, false, false);
    acc = __builtin_amdgcn_wmma_f32_16x16x32_bf16(false, a1, false, b1,
                                                  (short)0, acc, false, false);

    // ---- store D: VGPR r -> row (lanes 0-15: m0+r, lanes 16-31: m0+8+r), N=col
    const int mrow = m0 + hi * 8;
    float* orow = out + (long long)mrow * D_OUT + n0 + col;
    if (m0 + 16 <= n_nodes) {
        // Scalar-uniform full-tile fast path: 8 straight global_store_b32,
        // no per-lane EXEC juggling.
        #pragma unroll
        for (int r = 0; r < 8; ++r) {
            orow[(long long)r * D_OUT] = acc[r];
        }
    } else {
        // Tail tile (only possible for the last block when n_nodes % 16 != 0).
        #pragma unroll
        for (int r = 0; r < 8; ++r) {
            if (mrow + r < n_nodes) orow[(long long)r * D_OUT] = acc[r];
        }
    }
}

// ---------------------------------------------------------------------------
// Launch
// ---------------------------------------------------------------------------
extern "C" void kernel_launch(void* const* d_in, const int* in_sizes, int n_in,
                              void* d_out, int out_size, void* d_ws, size_t ws_size,
                              hipStream_t stream) {
    const float*     vertices = (const float*)d_in[0];
    const long long* edges    = (const long long*)d_in[1];
    const float*     weight   = (const float*)d_in[2];
    const float*     bias     = (const float*)d_in[3];
    float*           out      = (float*)d_out;

    const int n_nodes = in_sizes[0] / D_FEAT;
    const int n_edges = in_sizes[1] / 2;

    const size_t agg_elems = (size_t)n_nodes * D_FEAT;
    const bool ws_has_agg  = ws_size >= (agg_elems + (size_t)n_nodes) * sizeof(float);

    // agg: use scratch if it fits, else alias onto d_out (same shape; kernel 3
    // barriers all reads before any write, and blocks own disjoint rows).
    float* agg = ws_has_agg ? (float*)d_ws : out;
    float* deg = ws_has_agg ? ((float*)d_ws + agg_elems) : (float*)d_ws;

    // 1) zero accumulators (agg is a multiple of 4 floats: D_FEAT == 64)
    {
        long long n4 = (long long)(agg_elems / 4);
        int blocks = (int)((n4 + 255) / 256);
        if (blocks > 4096) blocks = 4096;
        gc_zero4_kernel<<<blocks, 256, 0, stream>>>((float4*)agg, n4);
        gc_zero_kernel<<<(n_nodes + 255) / 256, 256, 0, stream>>>(deg, n_nodes);
    }

    // 2) edge scatter: one wave per edge
    {
        long long threads = (long long)n_edges * 32;
        int blocks = (int)((threads + 255) / 256);
        gc_scatter_kernel<<<blocks, 256, 0, stream>>>(vertices, edges, agg, deg,
                                                      n_edges);
    }

    // 3) fused normalize + WMMA GEMM + bias
    {
        int blocks = (n_nodes + 15) / 16;
        gc_gemm_kernel<<<blocks, 128, 0, stream>>>(vertices, weight, bias,
                                                   agg, deg, out, n_nodes);
    }
}
